// DynamicTemperatureContrastiveLoss_8469675508135
// MI455X (gfx1250) — matrix operand order repaired and checked by reading, not compiled, not run on hardware
//
#include <hip/hip_runtime.h>
#include <hip/hip_bf16.h>

typedef _Float16 v16h __attribute__((ext_vector_type(16)));
typedef _Float16 v8h  __attribute__((ext_vector_type(8)));
typedef _Float16 v4h  __attribute__((ext_vector_type(4)));
typedef float    v8f  __attribute__((ext_vector_type(8)));

#define BN   4096
#define DN   768
#define TILE 128
#define KT   32
#define LDH  40   // LDS row stride in halves (32 + 8 pad -> no bank conflicts)

// ---------------------------------------------------------------------------
// Pass 1: L2-normalize V and T rows, emit f16 copies. One wave per row.
// ---------------------------------------------------------------------------
__global__ __launch_bounds__(256) void k_normalize(
    const float* __restrict__ V, const float* __restrict__ T,
    _Float16* __restrict__ Vn, _Float16* __restrict__ Tn)
{
  const int gw   = blockIdx.x * 8 + (threadIdx.x >> 5);   // 0 .. 2*BN-1
  const int lane = threadIdx.x & 31;
  const float* src; _Float16* dst; int row;
  if (gw < BN) { src = V; dst = Vn; row = gw; }
  else         { src = T; dst = Tn; row = gw - BN; }

  const float4* p = (const float4*)(src + (size_t)row * DN);
  float4 v[6];
  float ss = 0.f;
#pragma unroll
  for (int i = 0; i < 6; ++i) {
    v[i] = p[lane + 32 * i];
    ss += v[i].x*v[i].x + v[i].y*v[i].y + v[i].z*v[i].z + v[i].w*v[i].w;
  }
#pragma unroll
  for (int off = 16; off > 0; off >>= 1) ss += __shfl_xor(ss, off, 32);
  const float r = 1.0f / sqrtf(ss);

  _Float16* d = dst + (size_t)row * DN;
#pragma unroll
  for (int i = 0; i < 6; ++i) {
    v4h h;
    h[0] = (_Float16)(v[i].x * r);
    h[1] = (_Float16)(v[i].y * r);
    h[2] = (_Float16)(v[i].z * r);
    h[3] = (_Float16)(v[i].w * r);
    *(v4h*)(d + (size_t)(lane + 32 * i) * 4) = h;
  }
}

// ---------------------------------------------------------------------------
// Pass 2: histogram of match_ids (range [0,1024)) -> per-id counts and
// matched pair count  num_pos = sum(c^2)  (includes diagonal, as reference).
// ---------------------------------------------------------------------------
__global__ __launch_bounds__(1024) void k_hist(
    const int* __restrict__ ids, int* __restrict__ histOut, float* __restrict__ scal)
{
  __shared__ int h[1024];
  __shared__ int tot;
  const int tid = threadIdx.x;
  h[tid] = 0;
  if (tid == 0) tot = 0;
  __syncthreads();
  for (int i = tid; i < BN; i += 1024) atomicAdd(&h[ids[i] & 1023], 1);
  __syncthreads();
  const int c = h[tid];
  histOut[tid] = c;
  atomicAdd(&tot, c * c);
  __syncthreads();
  if (tid == 0) scal[7] = (float)tot;
}

// ---------------------------------------------------------------------------
// Zero the accumulator region of the workspace.
// ---------------------------------------------------------------------------
__global__ void k_init(float* __restrict__ z, int n) {
  const int i = blockIdx.x * blockDim.x + threadIdx.x;
  if (i < n) z[i] = 0.f;
}

// ---------------------------------------------------------------------------
// Pass 3: sim = Vn @ Tn^T via V_WMMA_F32_16X16X32_F16.
// 128x128 tile / block (256 thr = 8 waves). Wave owns 64x32 (4x2 wmma accs).
// Also accumulates total / matched / diagonal sums of sim for temp calc.
// ---------------------------------------------------------------------------
__global__ __launch_bounds__(256) void k_gemm(
    const _Float16* __restrict__ A, const _Float16* __restrict__ Bm,
    const int* __restrict__ ids, float* __restrict__ sim, float* __restrict__ scal)
{
  __shared__ __align__(16) _Float16 lA[TILE * LDH];
  __shared__ __align__(16) _Float16 lB[TILE * LDH];
  __shared__ int   rIds[TILE];
  __shared__ int   cIds[TILE];
  __shared__ float redT, redM, redD;

  const int tid     = threadIdx.x;
  const int rowBase = blockIdx.y * TILE;
  const int colBase = blockIdx.x * TILE;

  if (tid < TILE)            rIds[tid]        = ids[rowBase + tid];
  else if (tid < 2 * TILE)   cIds[tid - TILE] = ids[colBase + (tid - TILE)];
  if (tid == 0) { redT = 0.f; redM = 0.f; redD = 0.f; }

  const int lane  = tid & 31;
  const int wid   = tid >> 5;
  const int waveM = wid >> 2;   // 0..1  (64 rows each)
  const int waveN = wid & 3;    // 0..3  (32 cols each)

  const v8f vzero = {0.f,0.f,0.f,0.f,0.f,0.f,0.f,0.f};
  v8f acc[4][2];
#pragma unroll
  for (int mt = 0; mt < 4; ++mt)
#pragma unroll
    for (int nt = 0; nt < 2; ++nt) acc[mt][nt] = vzero;

  // global->LDS staging: 512 chunks of 8 halves per matrix; thread owns 2.
  const int r0 = tid >> 2;          // rows 0..63
  const int s0 = (tid & 3) * 8;     // k sub-offset 0/8/16/24
  const int r1 = r0 + 64;           // rows 64..127

  // ISA 7.12.2 fragment geometry (wave32):
  const int halfK = (lane < 16) ? 0 : 8;    // A: K base {0..7,16..23} / {8..15,24..31}
  const int bK    = (lane < 16) ? 0 : 8;    // B: K base {0..7,16..23} / {8..15,24..31}
  const int mrow  = lane & 15;

  for (int k0 = 0; k0 < DN; k0 += KT) {
    __syncthreads();
    *(v8h*)(lA + r0 * LDH + s0) = *(const v8h*)(A  + (size_t)(rowBase + r0) * DN + k0 + s0);
    *(v8h*)(lA + r1 * LDH + s0) = *(const v8h*)(A  + (size_t)(rowBase + r1) * DN + k0 + s0);
    *(v8h*)(lB + r0 * LDH + s0) = *(const v8h*)(Bm + (size_t)(colBase + r0) * DN + k0 + s0);
    *(v8h*)(lB + r1 * LDH + s0) = *(const v8h*)(Bm + (size_t)(colBase + r1) * DN + k0 + s0);
    if (k0 + KT < DN) {   // global_prefetch_b8 for the next K stage
      __builtin_prefetch(A  + (size_t)(rowBase + r0) * DN + k0 + KT + s0, 0, 1);
      __builtin_prefetch(Bm + (size_t)(colBase + r0) * DN + k0 + KT + s0, 0, 1);
    }
    __syncthreads();

    v16h afr[4];
#pragma unroll
    for (int mt = 0; mt < 4; ++mt) {
      const _Float16* p = lA + (size_t)(waveM * 64 + mt * 16 + mrow) * LDH + halfK;
      v8h lo = *(const v8h*)(p);
      v8h hi = *(const v8h*)(p + 16);
      afr[mt] = __builtin_shufflevector(lo, hi, 0,1,2,3,4,5,6,7,8,9,10,11,12,13,14,15);
    }
    v16h bfr[2];
#pragma unroll
    for (int nt = 0; nt < 2; ++nt) {
      const _Float16* p = lB + (size_t)(waveN * 32 + nt * 16 + mrow) * LDH + bK;
      v8h lo = *(const v8h*)(p);
      v8h hi = *(const v8h*)(p + 16);
      bfr[nt] = __builtin_shufflevector(lo, hi, 0,1,2,3,4,5,6,7,8,9,10,11,12,13,14,15);
    }
#pragma unroll
    for (int mt = 0; mt < 4; ++mt)
#pragma unroll
      for (int nt = 0; nt < 2; ++nt)
        acc[mt][nt] = __builtin_amdgcn_wmma_f32_16x16x32_f16(
            false, afr[mt], false, bfr[nt], (short)0, acc[mt][nt], false, false);
  }

  // Epilogue: spill sim tile + partial mask-weighted reductions.
  float tot = 0.f, mat = 0.f, dia = 0.f;
#pragma unroll
  for (int mt = 0; mt < 4; ++mt) {
#pragma unroll
    for (int nt = 0; nt < 2; ++nt) {
      const int N   = waveN * 32 + nt * 16 + mrow;
      const int Mb  = waveM * 64 + mt * 16 + ((lane < 16) ? 0 : 8);
      const int cid = cIds[N];
      const int gj  = colBase + N;
#pragma unroll
      for (int r = 0; r < 8; ++r) {
        const int   M  = Mb + r;
        const int   gi = rowBase + M;
        const float x  = acc[mt][nt][r];
        sim[(size_t)gi * BN + gj] = x;
        tot += x;
        if (rIds[M] == cid) mat += x;
        if (gi == gj)       dia += x;
      }
    }
  }
  atomicAdd(&redT, tot);
  atomicAdd(&redM, mat);
  atomicAdd(&redD, dia);
  __syncthreads();
  if (tid == 0) {
    atomicAdd(&scal[0], redT);
    atomicAdd(&scal[1], redM);
    atomicAdd(&scal[2], redD);
  }
}

// ---------------------------------------------------------------------------
// Pass 4: dynamic temperature (1 thread).
// ---------------------------------------------------------------------------
__global__ void k_temp(float* scal) {
  const float total = scal[0], matched = scal[1], trace = scal[2], mc = scal[7];
  const float posSum = matched - trace;
  const float posCnt = fmaxf(1.f, mc - (float)BN);
  const float negSum = total - matched;
  const float negCnt = fmaxf(1.f, (float)BN * (float)BN - mc);
  const float sep = posSum / posCnt - negSum / negCnt;
  float t = 0.07f * (0.8f + 0.4f * expf(-2.f * sep));
  t = fminf(0.2f, fmaxf(0.04f, t));
  scal[3] = t;
  scal[4] = 1.f / t;
}

// ---------------------------------------------------------------------------
// Pass 5: row logsumexp + matched-logit sum. Wave per row, sim from L2.
// |logits| <= 25 so plain sum-exp is f32-safe (no max shift needed).
// ---------------------------------------------------------------------------
__global__ __launch_bounds__(256) void k_rowlse(
    const float* __restrict__ sim, const int* __restrict__ ids,
    const int* __restrict__ hist, float* __restrict__ scal)
{
  const int row  = blockIdx.x * 8 + (threadIdx.x >> 5);
  const int lane = threadIdx.x & 31;
  const float invT = scal[4];
  const int myId = ids[row];
  const float* p = sim + (size_t)row * BN;
  float se = 0.f, ms = 0.f;
#pragma unroll 4
  for (int j = lane * 4; j < BN; j += 128) {
    const float4 x  = *(const float4*)(p + j);
    const int4  id4 = *(const int4*)(ids + j);
    se += __expf(x.x * invT) + __expf(x.y * invT)
        + __expf(x.z * invT) + __expf(x.w * invT);
    if (id4.x == myId) ms += x.x;
    if (id4.y == myId) ms += x.y;
    if (id4.z == myId) ms += x.z;
    if (id4.w == myId) ms += x.w;
  }
#pragma unroll
  for (int off = 16; off > 0; off >>= 1) {
    se += __shfl_xor(se, off, 32);
    ms += __shfl_xor(ms, off, 32);
  }
  if (lane == 0) {
    const float part = (float)hist[myId] * __logf(se) - ms * invT;
    atomicAdd(&scal[5], part);
  }
}

// ---------------------------------------------------------------------------
// Pass 6a: column partial sums (coalesced: lane = column). Grid (16,16).
// ---------------------------------------------------------------------------
__global__ __launch_bounds__(256) void k_colpart(
    const float* __restrict__ sim, const int* __restrict__ ids,
    const float* __restrict__ scal, float* __restrict__ colSE, float* __restrict__ colMS)
{
  const int j  = blockIdx.x * 256 + threadIdx.x;
  const int i0 = blockIdx.y * 256;
  const float invT = scal[4];
  const int myId = ids[j];
  float se = 0.f, ms = 0.f;
#pragma unroll 8
  for (int i = i0; i < i0 + 256; ++i) {
    const float x = sim[(size_t)i * BN + j];
    se += __expf(x * invT);
    if (ids[i] == myId) ms += x;
  }
  atomicAdd(&colSE[j], se);
  atomicAdd(&colMS[j], ms);
}

// Pass 6b: finish column LSE, accumulate t2v.
__global__ __launch_bounds__(256) void k_colfin(
    const int* __restrict__ ids, const int* __restrict__ hist,
    float* __restrict__ scal, const float* __restrict__ colSE,
    const float* __restrict__ colMS)
{
  __shared__ float red;
  if (threadIdx.x == 0) red = 0.f;
  __syncthreads();
  const int j = blockIdx.x * 256 + threadIdx.x;
  const float invT = scal[4];
  const float part = (float)hist[ids[j]] * __logf(colSE[j]) - colMS[j] * invT;
  atomicAdd(&red, part);
  __syncthreads();
  if (threadIdx.x == 0) atomicAdd(&scal[6], red);
}

// ---------------------------------------------------------------------------
// Pass 7: loss = (v2t + t2v) / (2 * num_pos)
// ---------------------------------------------------------------------------
__global__ void k_final(const float* scal, float* out) {
  const float np = fmaxf(1.f, scal[7]);
  out[0] = (scal[5] + scal[6]) / (2.f * np);
}

// ---------------------------------------------------------------------------
extern "C" void kernel_launch(void* const* d_in, const int* in_sizes, int n_in,
                              void* d_out, int out_size, void* d_ws, size_t ws_size,
                              hipStream_t stream)
{
  const float* V  = (const float*)d_in[0];
  const float* T  = (const float*)d_in[1];
  const int*  ids = (const int*)d_in[2];
  float* out = (float*)d_out;

  // Workspace layout:
  //   [Vn f16 BN*DN][Tn f16 BN*DN][sim f32 BN*BN][scal 32f][colSE 4096f][colMS 4096f][hist 1024i]
  _Float16* Vn  = (_Float16*)d_ws;
  _Float16* Tn  = Vn + (size_t)BN * DN;
  float*    sim = (float*)((char*)d_ws + (size_t)2 * BN * DN * sizeof(_Float16));
  float*    scal  = sim + (size_t)BN * BN;
  float*    colSE = scal + 32;
  float*    colMS = colSE + BN;
  int*      hist  = (int*)(colMS + BN);

  const int nz = 32 + 2 * BN;
  k_init     <<<(nz + 255) / 256, 256, 0, stream>>>(scal, nz);
  k_normalize<<<(2 * BN) / 8, 256, 0, stream>>>(V, T, Vn, Tn);
  k_hist     <<<1, 1024, 0, stream>>>(ids, hist, scal);
  k_gemm     <<<dim3(BN / TILE, BN / TILE), 256, 0, stream>>>(Vn, Tn, ids, sim, scal);
  k_temp     <<<1, 1, 0, stream>>>(scal);
  k_rowlse   <<<BN / 8, 256, 0, stream>>>(sim, ids, hist, scal);
  k_colpart  <<<dim3(16, 16), 256, 0, stream>>>(sim, ids, scal, colSE, colMS);
  k_colfin   <<<16, 256, 0, stream>>>(ids, hist, scal, colSE, colMS);
  k_final    <<<1, 1, 0, stream>>>(scal, out);
}
